// MPSUMambaEncoder_78675210928766
// MI455X (gfx1250) — matrix operand order repaired and checked by reading, not compiled, not run on hardware
//
#include <hip/hip_runtime.h>
#include <hip/hip_bf16.h>
#include <math.h>

// ---------------------------------------------------------------------------
// MPSU Mamba encoder forward for MI455X (gfx1250, wave32, WMMA).
// All GEMMs run through v_wmma_f32_16x16x32_bf16 (f32 accumulate).
// ---------------------------------------------------------------------------

#define BB   8
#define LL   2048
#define DD   256
#define DFF  1024
#define MM   (BB * LL)     // 16384 rows
#define CAP  128           // max captured nonzeros per adjacency row (~8 expected)
#define NCH  8             // scan chunks
#define CHL  (LL / NCH)    // 256 steps per chunk

typedef __attribute__((ext_vector_type(16))) __bf16 v16bf;
typedef __attribute__((ext_vector_type(8)))  __bf16 v8bf;
typedef __attribute__((ext_vector_type(8)))  float  v8f;

__device__ __forceinline__ __bf16 f2bf(float f) {
  // round-to-nearest-even f32 -> bf16
  unsigned u = __builtin_bit_cast(unsigned, f);
  unsigned r = (u + 0x7FFFu + ((u >> 16) & 1u)) >> 16;
  unsigned short s = (unsigned short)r;
  return __builtin_bit_cast(__bf16, s);
}

__device__ __forceinline__ v8f vzero8() {
  v8f z;
#pragma unroll
  for (int i = 0; i < 8; ++i) z[i] = 0.0f;
  return z;
}

__device__ __forceinline__ v8f wmma_bf16(v16bf a, v16bf b, v8f c) {
  // 8 args: (neg_a, A, neg_b, B, c_mod, C, reuse_a, reuse_b)
  return __builtin_amdgcn_wmma_f32_16x16x32_bf16(false, a, false, b, (short)0, c,
                                                 false, false);
}

// A fragment (16x32 bf16) from a row-major LDS tile.
// ISA layout: lane<16 -> row=lane, elems0-7=K(kb..kb+7), elems8-15=K(kb+16..23)
//             lane>=16 -> row=lane-16, K offsets +8 / +24.
__device__ __forceinline__ v16bf afrag_lds(const __bf16* rowPtr, int k32, int lane) {
  int o = k32 + ((lane & 16) ? 8 : 0);
  union { v16bf v; v8bf h[2]; } u;
  u.h[0] = *(const v8bf*)(rowPtr + o);
  u.h[1] = *(const v8bf*)(rowPtr + o + 16);
  return u.v;
}

// B fragment (32x16 bf16): col = lane%16, elems0-7=K(+8h..), elems8-15=K(16+8h..), h=lane/16.
__device__ __forceinline__ v16bf bfrag_lds(const __bf16* p /* &tile[k32][n0+lane%16] */, int ld, int lane) {
  int o = (lane & 16) ? 8 : 0;
  v16bf v;
#pragma unroll
  for (int i = 0; i < 8; ++i) { v[i] = p[(o + i) * ld]; v[8 + i] = p[(o + 16 + i) * ld]; }
  return v;
}

// B fragment straight from global f32 weights (converted in-register).
__device__ __forceinline__ v16bf bfrag_global_f32(const float* p /* &W[k32*ldw+n0+lane%16] */, int ld, int lane) {
  int o = (lane & 16) ? 8 : 0;
  v16bf v;
#pragma unroll
  for (int i = 0; i < 8; ++i) {
    v[i] = f2bf(p[(size_t)(o + i) * ld]);
    v[8 + i] = f2bf(p[(size_t)(o + 16 + i) * ld]);
  }
  return v;
}

// ---------------------------------------------------------------------------
// valid counts per batch: vc[b] = sum(mask[b,:])
// ---------------------------------------------------------------------------
__global__ void k_vc(const int* __restrict__ mask, int* __restrict__ vc) {
  __shared__ int red[256];
  int b = blockIdx.x, t = threadIdx.x;
  int s = 0;
  for (int l = t; l < LL; l += 256) s += mask[b * LL + l];
  red[t] = s; __syncthreads();
  for (int o = 128; o > 0; o >>= 1) {
    if (t < o) red[t] += red[t + o];
    __syncthreads();
  }
  if (t == 0) vc[b] = red[0];
}

// ---------------------------------------------------------------------------
// Deterministic per-row sparsity compaction of adjacency + degree.
// One block per (row i, batch b). Thread t owns columns [8t, 8t+8).
// ---------------------------------------------------------------------------
__global__ void k_csr(const float* __restrict__ A, int* __restrict__ csrJ,
                      float* __restrict__ csrV, int* __restrict__ cnt,
                      float* __restrict__ deg) {
  int i = blockIdx.x, b = blockIdx.y, t = threadIdx.x;
  const float* row = A + (size_t)(b * LL + i) * LL;
  __shared__ int   sc[256];
  __shared__ float rs[256];
  int   j0 = t * 8;
  float vals[8];
  int   c = 0;
  float ds = 0.f;
  __builtin_prefetch(row + j0, 0, 0);   // global_prefetch_b8
#pragma unroll
  for (int q = 0; q < 8; ++q) {
    float v = row[j0 + q];
    vals[q] = v; ds += v; c += (v != 0.f);
  }
  sc[t] = c; rs[t] = ds; __syncthreads();
  // inclusive Hillis-Steele scan (read-all / sync / write-all)
  for (int o = 1; o < 256; o <<= 1) {
    int v = (t >= o) ? sc[t - o] : 0;
    __syncthreads();
    sc[t] += v;
    __syncthreads();
  }
  int base = sc[t] - c;
  for (int o = 128; o > 0; o >>= 1) {
    if (t < o) rs[t] += rs[t + o];
    __syncthreads();
  }
  int rowid = b * LL + i;
  int w = 0;
#pragma unroll
  for (int q = 0; q < 8; ++q) {
    float v = vals[q];
    if (v != 0.f) {
      int p = base + w;
      if (p < CAP) { csrJ[(size_t)rowid * CAP + p] = j0 + q;
                     csrV[(size_t)rowid * CAP + p] = v; }
      ++w;
    }
  }
  if (t == 255) { int tot = sc[255]; cnt[rowid] = tot > CAP ? CAP : tot; }
  if (t == 0)   { float s = rs[0];   deg[rowid] = s < 1.f ? 1.f : s; }
}

// ---------------------------------------------------------------------------
// SpMM: out[b,i,:] = (sum_j A[b,i,j]*m[b,j]*Xin[b,j,:]) / deg[b,i]
// ---------------------------------------------------------------------------
__global__ void k_agg(const float* __restrict__ Xin, const int* __restrict__ mask,
                      const int* __restrict__ csrJ, const float* __restrict__ csrV,
                      const int* __restrict__ cnt, const float* __restrict__ deg,
                      float* __restrict__ out) {
  int i = blockIdx.x, b = blockIdx.y, d = threadIdx.x;
  int rowid = b * LL + i;
  int nn = cnt[rowid];
  const int*   J = csrJ + (size_t)rowid * CAP;
  const float* V = csrV + (size_t)rowid * CAP;
  float acc = 0.f;
  for (int p = 0; p < nn; ++p) {
    int j = J[p];
    float v = V[p] * (float)mask[b * LL + j];
    acc += v * Xin[(size_t)(b * LL + j) * DD + d];
  }
  out[(size_t)rowid * DD + d] = acc / deg[rowid];
}

// ---------------------------------------------------------------------------
// Row gather: out[b,l,:] = in[b, (l<vc[b] ? idx[b,l] : l), :]
// ---------------------------------------------------------------------------
__global__ void k_gather(const float* __restrict__ in, const int* __restrict__ idx,
                         const int* __restrict__ vc, float* __restrict__ out) {
  int q = blockIdx.x * 256 + threadIdx.x;     // over MM*DD
  int rowid = q >> 8;
  int d = q & 255;
  int b = rowid >> 11, l = rowid & (LL - 1);
  int sl = (l < vc[b]) ? idx[b * LL + l] : l;
  out[(size_t)q] = in[(size_t)(b * LL + sl) * DD + d];
}

// ---------------------------------------------------------------------------
// Chunked EMA scan: h_t = a*h_{t-1} + (1-a)*z_t  (z = zg[:, :, :256])
// pass1: per-chunk local scans; pass2: carry prefix with a^CHL; pass3: fix-up
// and fuse t = h * silu(g)   (g = zg[:, :, 256:]).
// ---------------------------------------------------------------------------
__global__ void k_scan1(const float* __restrict__ zg, const float* __restrict__ alog,
                        float* __restrict__ hloc, float* __restrict__ Hc) {
  int c = blockIdx.x, b = blockIdx.y, d = threadIdx.x;
  float a = 1.f / (1.f + expf(-alog[d]));
  float h = 0.f;
  for (int lt = 0; lt < CHL; ++lt) {
    int t = c * CHL + lt;
    float z = zg[(size_t)(b * LL + t) * 512 + d];
    h = a * h + (1.f - a) * z;
    hloc[(size_t)(b * LL + t) * DD + d] = h;
  }
  Hc[(size_t)(b * NCH + c) * DD + d] = h;
}

__global__ void k_scan2(const float* __restrict__ alog, const float* __restrict__ Hc,
                        float* __restrict__ Pb) {
  int b = blockIdx.x, d = threadIdx.x;
  float a = 1.f / (1.f + expf(-alog[d]));
  float aCL = a;
#pragma unroll
  for (int q = 0; q < 8; ++q) aCL *= aCL;   // a^256
  float P = 0.f;
  for (int c = 0; c < NCH; ++c) {
    Pb[(size_t)(b * NCH + c) * DD + d] = P;
    P = aCL * P + Hc[(size_t)(b * NCH + c) * DD + d];
  }
}

__global__ void k_scan3(const float* __restrict__ zg, const float* __restrict__ alog,
                        const float* __restrict__ Pb, float* __restrict__ hbuf) {
  int c = blockIdx.x, b = blockIdx.y, d = threadIdx.x;
  float a = 1.f / (1.f + expf(-alog[d]));
  float P = Pb[(size_t)(b * NCH + c) * DD + d];
  float p = a;
  for (int lt = 0; lt < CHL; ++lt) {
    int t = c * CHL + lt;
    size_t o = (size_t)(b * LL + t) * DD + d;
    float h = hbuf[o] + p * P;
    p *= a;
    float g = zg[(size_t)(b * LL + t) * 512 + 256 + d];
    float sg = g / (1.f + expf(-g));       // silu
    hbuf[o] = h * sg;
  }
}

// ---------------------------------------------------------------------------
// Generic bf16-WMMA GEMM:  Out[gm, nBase+0..63] (op)= post(A_row(gm) @ W)
//   A rows are 256-wide f32; optional gather+mask on A, bias/relu/mask/accum
//   on the epilogue. 64x64 block tile, 4 waves x (16 rows x 64 cols), K=256.
// ---------------------------------------------------------------------------
enum { GF_GATHER = 1, GF_MASKIN = 2, GF_RELU = 4, GF_MASKOUT = 8,
       GF_ACCUM = 16, GF_BIAS = 32 };

__global__ __launch_bounds__(128)
void k_gemm(const float* __restrict__ Asrc, int ldA,
            const float* __restrict__ W, int ldW,
            float* __restrict__ Out, int ldO,
            const float* __restrict__ bias,
            const int* __restrict__ gidx, const int* __restrict__ vc,
            const int* __restrict__ mask, int flags) {
  __shared__ __bf16 sA[64][256];   // 32 KB
  __shared__ __bf16 sW[256][64];   // 32 KB
  int tid = threadIdx.x, lane = tid & 31, wave = tid >> 5;
  int mBase = blockIdx.x * 64, nBase = blockIdx.y * 64;

  // stage A (64 rows x 256 K), converted to bf16, gather/mask applied
  for (int q = tid; q < 64 * 64; q += 128) {
    int r = q >> 6, c4 = q & 63;
    int gm = mBase + r;
    int b = gm >> 11, l = gm & (LL - 1);
    int sl = l;
    if (flags & GF_GATHER) { if (l < vc[b]) sl = gidx[b * LL + l]; }
    float mv = (flags & GF_MASKIN) ? (float)mask[b * LL + l] : 1.f;
    const float4 x = *(const float4*)(Asrc + (size_t)(b * LL + sl) * ldA + c4 * 4);
    sA[r][c4 * 4 + 0] = f2bf(x.x * mv);
    sA[r][c4 * 4 + 1] = f2bf(x.y * mv);
    sA[r][c4 * 4 + 2] = f2bf(x.z * mv);
    sA[r][c4 * 4 + 3] = f2bf(x.w * mv);
  }
  // stage W panel (256 K x 64 N)
  for (int q = tid; q < 256 * 16; q += 128) {
    int r = q >> 4, c4 = q & 15;
    const float4 x = *(const float4*)(W + (size_t)r * ldW + nBase + c4 * 4);
    sW[r][c4 * 4 + 0] = f2bf(x.x);
    sW[r][c4 * 4 + 1] = f2bf(x.y);
    sW[r][c4 * 4 + 2] = f2bf(x.z);
    sW[r][c4 * 4 + 3] = f2bf(x.w);
  }
  __syncthreads();

  v8f acc[4];
#pragma unroll
  for (int j = 0; j < 4; ++j) acc[j] = vzero8();
  int rB = wave * 16;
#pragma unroll
  for (int kb = 0; kb < 8; ++kb) {
    int k32 = kb * 32;
    v16bf af = afrag_lds(&sA[rB + (lane & 15)][0], k32, lane);
    int brow = k32;
#pragma unroll
    for (int j = 0; j < 4; ++j) {
      v16bf bf_ = bfrag_lds(&sW[brow][j * 16 + (lane & 15)], 64, lane);
      acc[j] = wmma_bf16(af, bf_, acc[j]);
    }
  }

  // epilogue: D layout -> VGPR e: lanes0-15 row e, lanes16-31 row e+8; col=lane%16
  int colLane = lane & 15;
  int rOff = (lane & 16) ? 8 : 0;
#pragma unroll
  for (int j = 0; j < 4; ++j) {
    int col = nBase + j * 16 + colLane;
    float bval = (flags & GF_BIAS) ? bias[col] : 0.f;
#pragma unroll
    for (int e = 0; e < 8; ++e) {
      int gm = mBase + rB + rOff + e;
      float v = acc[j][e] + bval;
      if (flags & GF_RELU) v = v > 0.f ? v : 0.f;
      if (flags & GF_MASKOUT) {
        int b = gm >> 11, l = gm & (LL - 1);
        v *= (float)mask[b * LL + l];
      }
      size_t oo = (size_t)gm * ldO + col;
      float prev = (flags & GF_ACCUM) ? Out[oo] : 0.f;
      Out[oo] = prev + v;
    }
  }
}

// ---------------------------------------------------------------------------
// AddNorm 1: X1 = X + LN(Y)*g + b   (one row per block, D=256)
// ---------------------------------------------------------------------------
__global__ void k_ln1(const float* __restrict__ X, const float* __restrict__ Y,
                      const float* __restrict__ g, const float* __restrict__ bb_,
                      float* __restrict__ X1) {
  __shared__ float r1[256], r2[256];
  int row = blockIdx.x, t = threadIdx.x;
  float y = Y[(size_t)row * DD + t];
  r1[t] = y; r2[t] = y * y; __syncthreads();
  for (int o = 128; o > 0; o >>= 1) {
    if (t < o) { r1[t] += r1[t + o]; r2[t] += r2[t + o]; }
    __syncthreads();
  }
  float mu = r1[0] / DD;
  float var = r2[0] / DD - mu * mu;
  float rv = rsqrtf(var + 1e-5f);
  X1[(size_t)row * DD + t] = X[(size_t)row * DD + t] + (y - mu) * rv * g[t] + bb_[t];
}

// ---------------------------------------------------------------------------
// Fused FFN + AddNorm2 per 16-row tile:
//   H = relu(X1@W1 + b1)  (bf16 in LDS, never hits HBM)
//   Y2 = H@W2 + b2 ; out = X1 + LN(Y2)*g2 + b2ln
// ---------------------------------------------------------------------------
__global__ __launch_bounds__(128)
void k_ffn(const float* __restrict__ X1, const float* __restrict__ W1,
           const float* __restrict__ b1, const float* __restrict__ W2,
           const float* __restrict__ b2, const float* __restrict__ g2,
           const float* __restrict__ bln2, float* __restrict__ out) {
  __shared__ __bf16 sX[16][264];    // 8.25 KB (stride 264 keeps 16B row alignment)
  __shared__ __bf16 sH[16][1032];   // 33 KB
  __shared__ float  sO[16][260];    // 16.6 KB
  __shared__ float  mus[16], rvs[16];
  int tid = threadIdx.x, lane = tid & 31, wave = tid >> 5;
  int mB = blockIdx.x * 16;
  int colLane = lane & 15;
  int rOff = (lane & 16) ? 8 : 0;

  for (int q = tid; q < 16 * 64; q += 128) {
    int r = q >> 6, c4 = q & 63;
    const float4 x = *(const float4*)(X1 + (size_t)(mB + r) * DD + c4 * 4);
    sX[r][c4 * 4 + 0] = f2bf(x.x);
    sX[r][c4 * 4 + 1] = f2bf(x.y);
    sX[r][c4 * 4 + 2] = f2bf(x.z);
    sX[r][c4 * 4 + 3] = f2bf(x.w);
  }
  __syncthreads();

  // GEMM1: 16 x 1024, K=256.  wave w owns cols [w*256, w*256+256)
  for (int jt = 0; jt < 16; ++jt) {
    int n0 = wave * 256 + jt * 16;
    v8f acc = vzero8();
#pragma unroll
    for (int kb = 0; kb < 8; ++kb) {
      int k32 = kb * 32;
      v16bf af = afrag_lds(&sX[lane & 15][0], k32, lane);
      v16bf bf_ = bfrag_global_f32(W1 + (size_t)k32 * DFF + n0 + colLane, DFF, lane);
      acc = wmma_bf16(af, bf_, acc);
    }
    int col = n0 + colLane;
    float bb1 = b1[col];
#pragma unroll
    for (int e = 0; e < 8; ++e) {
      float v = acc[e] + bb1;
      v = v > 0.f ? v : 0.f;
      sH[rOff + e][col] = f2bf(v);
    }
  }
  __syncthreads();

  // GEMM2: 16 x 256, K=1024. wave w owns cols [w*64, w*64+64)
  for (int jt = 0; jt < 4; ++jt) {
    int n0 = wave * 64 + jt * 16;
    v8f acc = vzero8();
#pragma unroll
    for (int kb = 0; kb < 32; ++kb) {
      int k32 = kb * 32;
      v16bf af = afrag_lds(&sH[lane & 15][0], k32, lane);
      v16bf bf_ = bfrag_global_f32(W2 + (size_t)k32 * DD + n0 + colLane, DD, lane);
      acc = wmma_bf16(af, bf_, acc);
    }
    int col = n0 + colLane;
    float bb2 = b2[col];
#pragma unroll
    for (int e = 0; e < 8; ++e) sO[rOff + e][col] = acc[e] + bb2;
  }
  __syncthreads();

  if (tid < 16) {
    float s = 0.f, s2 = 0.f;
    for (int c = 0; c < DD; ++c) { float v = sO[tid][c]; s += v; s2 += v * v; }
    float mu = s / DD;
    float var = s2 / DD - mu * mu;
    mus[tid] = mu;
    rvs[tid] = rsqrtf(var + 1e-5f);
  }
  __syncthreads();

  for (int q = tid; q < 16 * DD; q += 128) {
    int r = q >> 8, c = q & 255;
    float y = sO[r][c];
    size_t oo = (size_t)(mB + r) * DD + c;
    out[oo] = X1[oo] + (y - mus[r]) * rvs[r] * g2[c] + bln2[c];
  }
}

// ---------------------------------------------------------------------------
// Host-side orchestration
// ---------------------------------------------------------------------------
extern "C" void kernel_launch(void* const* d_in, const int* in_sizes, int n_in,
                              void* d_out, int out_size, void* d_ws, size_t ws_size,
                              hipStream_t stream) {
  (void)in_sizes; (void)n_in; (void)out_size; (void)ws_size;
  const float* X        = (const float*)d_in[0];
  const int*   nm       = (const int*)d_in[1];
  const float* Adj      = (const float*)d_in[2];
  const int*   si[3]    = {(const int*)d_in[3], (const int*)d_in[5], (const int*)d_in[7]};
  const int*   ri[3]    = {(const int*)d_in[4], (const int*)d_in[6], (const int*)d_in[8]};
  const float* gm1_Win  = (const float*)d_in[9];
  const float* gm1_al   = (const float*)d_in[10];
  const float* gm1_Wout = (const float*)d_in[11];
  const float* gm2_We   = (const float*)d_in[12];
  const float* gm_Win   = (const float*)d_in[13];
  const float* gm_al    = (const float*)d_in[14];
  const float* gm_Wout  = (const float*)d_in[15];
  const float* gm_We    = (const float*)d_in[16];
  const float* ln1g     = (const float*)d_in[17];
  const float* ln1b     = (const float*)d_in[18];
  const float* ln2g     = (const float*)d_in[19];
  const float* ln2b     = (const float*)d_in[20];
  const float* W1       = (const float*)d_in[21];
  const float* b1       = (const float*)d_in[22];
  const float* W2       = (const float*)d_in[23];
  const float* b2       = (const float*)d_in[24];
  const float* Wv       = (const float*)d_in[25];
  const float* bv       = (const float*)d_in[26];
  float* out = (float*)d_out;

  // workspace carve (floats; every region is a multiple of 64 floats -> 16B ok)
  float* ws = (float*)d_ws;
  size_t o = 0;
  float* bufZG = ws + o; o += (size_t)MM * 512;     // zg [M,512]; reused for RE
  float* bufT  = ws + o; o += (size_t)MM * 256;     // h/t, then agg
  float* bufBO = ws + o; o += (size_t)MM * 256;     // bo, then out_k / branch3
  float* bufY  = ws + o; o += (size_t)MM * 256;     // Y accumulator
  float* bufX1 = ws + o; o += (size_t)MM * 256;     // X1
  float* csrV  = ws + o; o += (size_t)MM * CAP;
  int*   csrJ  = (int*)(ws + o); o += (size_t)MM * CAP;
  int*   csrC  = (int*)(ws + o); o += MM;
  float* degA  = ws + o; o += MM;
  int*   vcA   = (int*)(ws + o); o += 64;
  float* Hc    = ws + o; o += (size_t)BB * NCH * DD;
  float* Pb    = ws + o; o += (size_t)BB * NCH * DD;
  float* bufRE = bufZG;   // zg is dead once the scan finishes

  dim3 gRows(MM / 64, 512 / 64);   // N=512 GEMMs
  dim3 gRowsD(MM / 64, DD / 64);   // N=256 GEMMs
  dim3 gBL(LL, BB);
  dim3 gScan(NCH, BB);

  k_vc<<<BB, 256, 0, stream>>>(nm, vcA);
  k_csr<<<gBL, 256, 0, stream>>>(Adj, csrJ, csrV, csrC, degA);

  for (int k = 0; k < 3; ++k) {
    // zg = (gather(X, si)*m) @ Win
    k_gemm<<<gRows, 128, 0, stream>>>(X, DD, gm1_Win, 512, bufZG, 512,
                                      nullptr, si[k], vcA, nm, GF_GATHER | GF_MASKIN);
    k_scan1<<<gScan, 256, 0, stream>>>(bufZG, gm1_al, bufT, Hc);
    k_scan2<<<BB, 256, 0, stream>>>(gm1_al, Hc, Pb);
    k_scan3<<<gScan, 256, 0, stream>>>(bufZG, gm1_al, Pb, bufT);
    // bo = (t @ Wout) * m
    k_gemm<<<gRowsD, 128, 0, stream>>>(bufT, DD, gm1_Wout, DD, bufBO, DD,
                                       nullptr, nullptr, vcA, nm, GF_MASKOUT);
    // re = gather(bo, ri)
    k_gather<<<MM * DD / 256, 256, 0, stream>>>(bufBO, ri[k], vcA, bufRE);
    // agg = (A @ (re*m)) / deg
    k_agg<<<gBL, 256, 0, stream>>>(bufRE, nm, csrJ, csrV, csrC, degA, bufT);
    // out_k = relu(agg @ We) * m
    k_gemm<<<gRowsD, 128, 0, stream>>>(bufT, DD, gm2_We, DD, bufBO, DD,
                                       nullptr, nullptr, vcA, nm, GF_RELU | GF_MASKOUT);
    // Y (+)= out_k @ Wv_k  (+ bv on first)
    k_gemm<<<gRowsD, 128, 0, stream>>>(bufBO, DD, Wv + (size_t)k * DD * DD, DD,
                                       bufY, DD, bv, nullptr, vcA, nm,
                                       (k == 0) ? GF_BIAS : GF_ACCUM);
  }

  // direct branch: mamba(X) + graph_agg(X)
  k_gemm<<<gRows, 128, 0, stream>>>(X, DD, gm_Win, 512, bufZG, 512,
                                    nullptr, nullptr, vcA, nm, GF_MASKIN);
  k_scan1<<<gScan, 256, 0, stream>>>(bufZG, gm_al, bufT, Hc);
  k_scan2<<<BB, 256, 0, stream>>>(gm_al, Hc, Pb);
  k_scan3<<<gScan, 256, 0, stream>>>(bufZG, gm_al, Pb, bufT);
  k_gemm<<<gRowsD, 128, 0, stream>>>(bufT, DD, gm_Wout, DD, bufBO, DD,
                                     nullptr, nullptr, vcA, nm, GF_MASKOUT);
  k_agg<<<gBL, 256, 0, stream>>>(X, nm, csrJ, csrV, csrC, degA, bufT);
  // bufBO = bo + relu(agg @ We) * m
  k_gemm<<<gRowsD, 128, 0, stream>>>(bufT, DD, gm_We, DD, bufBO, DD,
                                     nullptr, nullptr, vcA, nm,
                                     GF_RELU | GF_MASKOUT | GF_ACCUM);
  k_gemm<<<gRowsD, 128, 0, stream>>>(bufBO, DD, Wv + (size_t)3 * DD * DD, DD,
                                     bufY, DD, nullptr, nullptr, vcA, nm, GF_ACCUM);

  k_ln1<<<MM, 256, 0, stream>>>(X, bufY, ln1g, ln1b, bufX1);
  k_ffn<<<MM / 16, 128, 0, stream>>>(bufX1, W1, b1, W2, b2, ln2g, ln2b, out);
}
